// WindowAttention_59931973648811
// MI455X (gfx1250) — matrix operand order repaired and checked by reading, not compiled, not run on hardware
//
#include <hip/hip_runtime.h>
#include <hip/hip_bf16.h>
#include <stdint.h>

// ---------------------------------------------------------------------------
// Swin window attention, fused into 3 WMMA kernels for gfx1250 (wave32).
//   B_=4096 windows, N=49 tokens (pad->64), DIM=384, H=12 heads, hd=32, NW=64
// All GEMMs run on v_wmma_f32_16x16x32_bf16 (16x16 tiles, K=32).
// Attention Q/K tiles are staged with global_load_async_to_lds_b128 (ASYNCcnt).
// ---------------------------------------------------------------------------

typedef __attribute__((ext_vector_type(16))) __bf16 v16bf;   // 8 VGPRs
typedef __attribute__((ext_vector_type(8)))  float  v8f;     // 8 VGPRs

union Frag16 { v16bf v; unsigned int u[8]; };

__device__ __forceinline__ unsigned short f32_to_bf16(float f) {
    unsigned int u = __float_as_uint(f);
    u += 0x7FFFu + ((u >> 16) & 1u);          // round-to-nearest-even
    return (unsigned short)(u >> 16);
}
__device__ __forceinline__ unsigned int pack_bf16(float a, float b) {
    return (unsigned int)f32_to_bf16(a) | ((unsigned int)f32_to_bf16(b) << 16);
}

#define HEADS   12
#define HD      32
#define NTOK    49
#define DIM     384
#define NQKV    1152
#define MTOTAL  200704          // 4096 * 49
#define NWIN    64

// ===========================================================================
// Kernel 1: qkv = x @ qkv_w^T + qkv_b  -> bf16 Q(scaled)/K/V in (b,h,n,d)
//   Tile: 128(M) x 64(N) per 256-thread block, K stepped by 32.
// ===========================================================================
__global__ __launch_bounds__(256)
void qkv_gemm_kernel(const float* __restrict__ x,
                     const float* __restrict__ qkv_w,
                     const float* __restrict__ qkv_b,
                     unsigned short* __restrict__ q_ws,
                     unsigned short* __restrict__ k_ws,
                     unsigned short* __restrict__ v_ws)
{
    __shared__ unsigned int Asu[128][17];   // 128 rows x 32 k (bf16 pairs), +pad
    __shared__ unsigned int Bsu[64][17];    // 64 n-rows x 32 k (bf16 pairs)

    const int t    = threadIdx.x;
    const int lane = t & 31;
    const int wave = t >> 5;                // 8 waves
    const int wm   = wave >> 1;             // 0..3  -> m sub-offset 32*wm
    const int wn   = wave & 1;              // 0..1  -> n sub-offset 32*wn
    const int m0   = blockIdx.x * 128;
    const int n0   = blockIdx.y * 64;
    const int h    = lane >> 4;             // half-wave selector
    const int ml   = lane & 15;             // M (A) / N (B) lane index

    v8f acc[2][2];
    #pragma unroll
    for (int i = 0; i < 2; ++i)
        #pragma unroll
        for (int j = 0; j < 2; ++j)
            #pragma unroll
            for (int r = 0; r < 8; ++r) acc[i][j][r] = 0.0f;

    for (int k0 = 0; k0 < DIM; k0 += 32) {
        // stage A tile (fp32 -> bf16 pairs)
        for (int w = t; w < 128 * 16; w += 256) {
            int row = w >> 4, cu = w & 15;
            const float* ap = x + (size_t)(m0 + row) * DIM + k0 + cu * 2;
            Asu[row][cu] = pack_bf16(ap[0], ap[1]);
        }
        // stage B tile: B[k][n] = qkv_w[n][k]
        for (int w = t; w < 64 * 16; w += 256) {
            int row = w >> 4, cu = w & 15;
            const float* bp = qkv_w + (size_t)(n0 + row) * DIM + k0 + cu * 2;
            Bsu[row][cu] = pack_bf16(bp[0], bp[1]);
        }
        if (k0 + 32 < DIM)  // speculative prefetch of next A chunk
            __builtin_prefetch(x + (size_t)(m0 + (t >> 1)) * DIM + k0 + 32, 0, 0);
        __syncthreads();

        #pragma unroll
        for (int i = 0; i < 2; ++i) {
            Frag16 a;
            const unsigned int* ar = &Asu[wm * 32 + i * 16 + ml][0];
            #pragma unroll
            for (int g = 0; g < 8; ++g)
                a.u[g] = ar[h * 8 + g];
            #pragma unroll
            for (int j = 0; j < 2; ++j) {
                Frag16 bf;
                const unsigned int* br = &Bsu[wn * 32 + j * 16 + ml][0];
                #pragma unroll
                for (int g = 0; g < 8; ++g) bf.u[g] = br[h * 8 + g];
                acc[i][j] = __builtin_amdgcn_wmma_f32_16x16x32_bf16(
                    false, a.v, false, bf.v, (short)0, acc[i][j], false, false);
            }
        }
        __syncthreads();
    }

    // epilogue: +bias, scale q, scatter bf16 into (b, head, token, d)
    const float scale = 0.17677669529663687f;   // 32^-0.5
    #pragma unroll
    for (int i = 0; i < 2; ++i) {
        #pragma unroll
        for (int j = 0; j < 2; ++j) {
            #pragma unroll
            for (int r = 0; r < 8; ++r) {
                int m = m0 + wm * 32 + i * 16 + r + 8 * h;
                int n = n0 + wn * 32 + j * 16 + ml;
                float v = acc[i][j][r] + qkv_b[n];
                int part = n / DIM;             // 0=q 1=k 2=v (constant per block)
                int cc   = n - part * DIM;
                int head = cc >> 5, d = cc & 31;
                int b    = m / NTOK, tok = m - b * NTOK;
                size_t di = (((size_t)b * HEADS + head) * NTOK + tok) * HD + d;
                if (part == 0)      q_ws[di] = f32_to_bf16(v * scale);
                else if (part == 1) k_ws[di] = f32_to_bf16(v);
                else                v_ws[di] = f32_to_bf16(v);
            }
        }
    }
}

// ===========================================================================
// Kernel 2: per-(window,head) attention. One wave per block.
//   S = Q K^T (+rpb +mask), softmax rows, O = P V.  49 padded to 64.
//   Q/K tiles staged via async global->LDS (ASYNCcnt), 16B-aligned rows.
// ===========================================================================
__global__ __launch_bounds__(32)
void attn_kernel(const unsigned short* __restrict__ q_ws,
                 const unsigned short* __restrict__ k_ws,
                 const unsigned short* __restrict__ v_ws,
                 const float* __restrict__ attn_mask,
                 const float* __restrict__ rpb_table,
                 unsigned short* __restrict__ o_ws)
{
    // Row stride 20 uints = 80B: 16B-aligned for async B128 stores, and
    // 20*row mod 64 covers 16 distinct banks -> conflict-free frag gathers.
    __shared__ __attribute__((aligned(16))) unsigned int Qs[64][20];
    __shared__ __attribute__((aligned(16))) unsigned int Ks[64][20];
    __shared__ unsigned short Vt[32][66];    // d-major (transposed) for B frags
    __shared__ float          S[64][65];     // fp32 logits / exp scratch
    __shared__ unsigned int   P[64][33];     // softmax probs, bf16 pairs (64 j)

    const int l  = threadIdx.x;
    const int bh = blockIdx.x;
    const int b  = bh / HEADS, hh = bh - b * HEADS;
    const int h  = l >> 4, ml = l & 15;

    const size_t base = (size_t)bh * NTOK * HD;

    // ---- async-stage Q, K: 49 rows x 64B = 196 chunks of 16B each ----
    {
        const unsigned qlds = (unsigned)(uintptr_t)(&Qs[0][0]);
        const unsigned klds = (unsigned)(uintptr_t)(&Ks[0][0]);
        const unsigned long long qga = (unsigned long long)(q_ws + base);
        const unsigned long long kga = (unsigned long long)(k_ws + base);
        for (int idx = l; idx < NTOK * 4; idx += 32) {
            int row = idx >> 2, seg = idx & 3;
            unsigned loff = (unsigned)(row * 80 + seg * 16);
            unsigned goff = (unsigned)(idx * 16);
            unsigned lq = qlds + loff;
            unsigned lk = klds + loff;
            asm volatile("global_load_async_to_lds_b128 %0, %1, %2"
                         :: "v"(lq), "v"(goff), "s"(qga) : "memory");
            asm volatile("global_load_async_to_lds_b128 %0, %1, %2"
                         :: "v"(lk), "v"(goff), "s"(kga) : "memory");
        }
    }
    // ---- zero-pad Q/K rows 49..63 (only the 16 data words matter) ----
    for (int w = l; w < 15 * 16; w += 32) {
        int row = NTOK + (w >> 4), cu = w & 15;
        Qs[row][cu] = 0u;
        Ks[row][cu] = 0u;
    }
    // ---- stage V transposed: Vt[d][token] (32-bit global reads) ----
    {
        const unsigned int* vg = (const unsigned int*)(v_ws + base);
        for (int w = l; w < NTOK * 16; w += 32) {
            int tok = w >> 4, du = w & 15;
            unsigned int vv = vg[w];
            Vt[2 * du][tok]     = (unsigned short)(vv & 0xffffu);
            Vt[2 * du + 1][tok] = (unsigned short)(vv >> 16);
        }
        for (int w = l; w < 32 * 15; w += 32) {
            int d = w / 15, tok = NTOK + (w % 15);
            Vt[d][tok] = 0;
        }
    }
    asm volatile("s_wait_asynccnt 0x0" ::: "memory");
    __syncthreads();

    // ---- S = Q K^T, fused rel-pos bias + window mask ----
    const float* mask = attn_mask + (size_t)(b & (NWIN - 1)) * NTOK * NTOK;
    #pragma unroll
    for (int mt = 0; mt < 4; ++mt) {
        Frag16 a;
        const unsigned int* ar = &Qs[mt * 16 + ml][0];
        #pragma unroll
        for (int g = 0; g < 8; ++g)
            a.u[g] = ar[h * 8 + g];
        #pragma unroll
        for (int nt = 0; nt < 4; ++nt) {
            Frag16 bf;
            const unsigned int* br = &Ks[nt * 16 + ml][0];
            #pragma unroll
            for (int g = 0; g < 8; ++g) bf.u[g] = br[h * 8 + g];
            v8f c;
            #pragma unroll
            for (int r = 0; r < 8; ++r) c[r] = 0.0f;
            c = __builtin_amdgcn_wmma_f32_16x16x32_bf16(
                false, a.v, false, bf.v, (short)0, c, false, false);
            #pragma unroll
            for (int r = 0; r < 8; ++r) {
                int i = mt * 16 + r + 8 * h;
                int j = nt * 16 + ml;
                float val = -1e30f;
                if (i < NTOK && j < NTOK) {
                    int yi = i / 7, xi = i - yi * 7;
                    int yj = j / 7, xj = j - yj * 7;
                    int ridx = (yi - yj + 6) * 13 + (xi - xj + 6);
                    val = c[r] + rpb_table[ridx * HEADS + hh] + mask[i * NTOK + j];
                }
                S[i][j] = val;
            }
        }
    }
    __syncthreads();

    // ---- row softmax (2 rows per lane), write bf16 P ----
    #pragma unroll
    for (int rr = 0; rr < 2; ++rr) {
        int row = l * 2 + rr;
        float mx = -3.0e38f;
        for (int j = 0; j < 64; ++j) mx = fmaxf(mx, S[row][j]);
        float sum = 0.0f;
        for (int j = 0; j < 64; ++j) {
            float e = __expf(S[row][j] - mx);
            S[row][j] = e;
            sum += e;
        }
        float inv = 1.0f / sum;
        for (int ju = 0; ju < 32; ++ju)
            P[row][ju] = pack_bf16(S[row][2 * ju] * inv, S[row][2 * ju + 1] * inv);
    }
    __syncthreads();

    // ---- O = P V  (K=64 -> two k-chunks), scatter bf16 to (b, n, c) ----
    #pragma unroll
    for (int mt = 0; mt < 4; ++mt) {
        #pragma unroll
        for (int nt = 0; nt < 2; ++nt) {
            v8f c;
            #pragma unroll
            for (int r = 0; r < 8; ++r) c[r] = 0.0f;
            #pragma unroll
            for (int kt = 0; kt < 2; ++kt) {
                Frag16 a;
                const unsigned int* pr = &P[mt * 16 + ml][kt * 16];
                #pragma unroll
                for (int g = 0; g < 8; ++g)
                    a.u[g] = pr[h * 8 + g];
                Frag16 bf;
                int d = nt * 16 + ml;
                #pragma unroll
                for (int g = 0; g < 8; ++g)
                    bf.u[g] = *(const unsigned int*)&Vt[d][kt * 32 + h * 16 + 2 * g];
                c = __builtin_amdgcn_wmma_f32_16x16x32_bf16(
                    false, a.v, false, bf.v, (short)0, c, false, false);
            }
            #pragma unroll
            for (int r = 0; r < 8; ++r) {
                int tok = mt * 16 + r + 8 * h;
                if (tok < NTOK) {
                    int d = nt * 16 + ml;
                    o_ws[((size_t)b * NTOK + tok) * DIM + hh * HD + d] =
                        f32_to_bf16(c[r]);
                }
            }
        }
    }
}

// ===========================================================================
// Kernel 3: out = attn_out @ proj_w^T + proj_b   (fp32 result)
// ===========================================================================
__global__ __launch_bounds__(256)
void proj_gemm_kernel(const unsigned short* __restrict__ o_ws,
                      const float* __restrict__ proj_w,
                      const float* __restrict__ proj_b,
                      float* __restrict__ out)
{
    __shared__ unsigned int Asu[128][17];
    __shared__ unsigned int Bsu[64][17];

    const int t    = threadIdx.x;
    const int lane = t & 31;
    const int wave = t >> 5;
    const int wm   = wave >> 1;
    const int wn   = wave & 1;
    const int m0   = blockIdx.x * 128;
    const int n0   = blockIdx.y * 64;
    const int h    = lane >> 4;
    const int ml   = lane & 15;

    const unsigned int* ag = (const unsigned int*)o_ws;   // rows of 192 uints

    v8f acc[2][2];
    #pragma unroll
    for (int i = 0; i < 2; ++i)
        #pragma unroll
        for (int j = 0; j < 2; ++j)
            #pragma unroll
            for (int r = 0; r < 8; ++r) acc[i][j][r] = 0.0f;

    for (int k0 = 0; k0 < DIM; k0 += 32) {
        for (int w = t; w < 128 * 16; w += 256) {
            int row = w >> 4, cu = w & 15;
            Asu[row][cu] = ag[(size_t)(m0 + row) * (DIM / 2) + (k0 >> 1) + cu];
        }
        for (int w = t; w < 64 * 16; w += 256) {
            int row = w >> 4, cu = w & 15;
            const float* bp = proj_w + (size_t)(n0 + row) * DIM + k0 + cu * 2;
            Bsu[row][cu] = pack_bf16(bp[0], bp[1]);
        }
        __syncthreads();

        #pragma unroll
        for (int i = 0; i < 2; ++i) {
            Frag16 a;
            const unsigned int* ar = &Asu[wm * 32 + i * 16 + ml][0];
            #pragma unroll
            for (int g = 0; g < 8; ++g)
                a.u[g] = ar[h * 8 + g];
            #pragma unroll
            for (int j = 0; j < 2; ++j) {
                Frag16 bf;
                const unsigned int* br = &Bsu[wn * 32 + j * 16 + ml][0];
                #pragma unroll
                for (int g = 0; g < 8; ++g) bf.u[g] = br[h * 8 + g];
                acc[i][j] = __builtin_amdgcn_wmma_f32_16x16x32_bf16(
                    false, a.v, false, bf.v, (short)0, acc[i][j], false, false);
            }
        }
        __syncthreads();
    }

    #pragma unroll
    for (int i = 0; i < 2; ++i)
        #pragma unroll
        for (int j = 0; j < 2; ++j)
            #pragma unroll
            for (int r = 0; r < 8; ++r) {
                int m = m0 + wm * 32 + i * 16 + r + 8 * h;
                int n = n0 + wn * 32 + j * 16 + ml;
                out[(size_t)m * DIM + n] = acc[i][j][r] + proj_b[n];
            }
}

// ===========================================================================
// Host-side launch
// inputs: x, attn_mask, qkv_w, qkv_b, proj_w, proj_b, rpb_table  (all fp32)
// ===========================================================================
extern "C" void kernel_launch(void* const* d_in, const int* in_sizes, int n_in,
                              void* d_out, int out_size, void* d_ws, size_t ws_size,
                              hipStream_t stream)
{
    (void)in_sizes; (void)n_in; (void)out_size; (void)ws_size;

    const float* x         = (const float*)d_in[0];
    const float* attn_mask = (const float*)d_in[1];
    const float* qkv_w     = (const float*)d_in[2];
    const float* qkv_b     = (const float*)d_in[3];
    const float* proj_w    = (const float*)d_in[4];
    const float* proj_b    = (const float*)d_in[5];
    const float* rpb_table = (const float*)d_in[6];
    float*       out       = (float*)d_out;

    // bf16 staging buffers: Q, K, V, attn-out (each 200704 x 384 bf16)
    const size_t per = (size_t)MTOTAL * DIM * sizeof(unsigned short);
    char* p = (char*)d_ws;
    unsigned short* q_ws = (unsigned short*)(p);
    unsigned short* k_ws = (unsigned short*)(p + per);
    unsigned short* v_ws = (unsigned short*)(p + 2 * per);
    unsigned short* o_ws = (unsigned short*)(p + 3 * per);

    // 1) QKV projection: M=200704 (1568 x 128), N=1152 (18 x 64)
    qkv_gemm_kernel<<<dim3(MTOTAL / 128, NQKV / 64), 256, 0, stream>>>(
        x, qkv_w, qkv_b, q_ws, k_ws, v_ws);

    // 2) attention: one wave per (window, head)
    attn_kernel<<<dim3(4096 * HEADS), 32, 0, stream>>>(
        q_ws, k_ws, v_ws, attn_mask, rpb_table, o_ws);

    // 3) output projection: N=384 (6 x 64)
    proj_gemm_kernel<<<dim3(MTOTAL / 128, DIM / 64), 256, 0, stream>>>(
        o_ws, proj_w, proj_b, out);
}
